// ClusteringAffinity_3556232921452
// MI455X (gfx1250) — matrix operand order repaired and checked by reading, not compiled, not run on hardware
//
#include <hip/hip_runtime.h>
#include <stdint.h>

// ---------------- problem sizes (compile-time) ----------------
#define HID    512      // hidden
#define NBATCH 512      // batch
#define MC     4000     // n_classes * n_centers
#define NT2_M  16       // NBATCH/32 super-tiles
#define NT2_N  125      // MC/32 super-tiles
#define OUTC   1001     // n_classes + 1

typedef __attribute__((ext_vector_type(8)))  __bf16 v8bf;
typedef __attribute__((ext_vector_type(16))) __bf16 v16bf;
typedef __attribute__((ext_vector_type(8)))  float  v8f;

#define WMMA_BF16(A, B, C) \
  __builtin_amdgcn_wmma_f32_16x16x32_bf16(false, (A), false, (B), (short)0, (C), false, false)

// =====================================================================
// 1) split f32 row-major [rows, 512] into bf16 hi/lo planes + row ||x||^2
//    hi = top 16 bits of f32 (exact residual lo = x - hi)
// =====================================================================
__global__ __launch_bounds__(256) void split_kernel(const float* __restrict__ src,
                                                    unsigned short* __restrict__ hi,
                                                    unsigned short* __restrict__ lo,
                                                    float* __restrict__ sq) {
  const int row = blockIdx.x;
  const int t = threadIdx.x;
  const float* p = src + (size_t)row * HID;
  float ss = 0.f;
  #pragma unroll
  for (int i = t; i < HID; i += 256) {
    float x = p[i];
    unsigned bx = __float_as_uint(x);
    unsigned hb = bx & 0xFFFF0000u;
    float hf = __uint_as_float(hb);
    float r = x - hf;                       // exact (mantissa prefix cancel)
    hi[(size_t)row * HID + i] = (unsigned short)(hb >> 16);
    lo[(size_t)row * HID + i] = (unsigned short)(__float_as_uint(r) >> 16);
    ss += x * x;
  }
  #pragma unroll
  for (int o = 16; o > 0; o >>= 1) ss += __shfl_xor(ss, o);
  __shared__ float red[8];
  if ((t & 31) == 0) red[t >> 5] = ss;
  __syncthreads();
  if (t == 0) {
    float tot = 0.f;
    #pragma unroll
    for (int i = 0; i < 8; i++) tot += red[i];
    sq[row] = tot;
  }
}

// =====================================================================
// WMMA tile loaders (row-major source, ld = HID)
// A 16x32 bf16: lane m = lane&15; lanes 0-15 hold K {0..7,16..23},
//               lanes 16-31 hold K {8..15,24..31}  -> two 8-elem chunks
// B 32x16 bf16: lane n = lane&15; lanes 0-15 hold K {0..7,16..23},
//               lanes 16-31 hold K {8..15,24..31}  -> two 8-elem chunks
// =====================================================================
__device__ __forceinline__ v16bf load_tile_a(const unsigned short* __restrict__ base,
                                             int row_base, int k, int lane) {
  const int r  = row_base + (lane & 15);
  const int kc = k + ((lane >> 4) << 3);
  const __bf16* q = (const __bf16*)(base + (size_t)r * HID + kc);
  v8bf c0 = *(const v8bf*)(q);
  v8bf c1 = *(const v8bf*)(q + 16);
  return __builtin_shufflevector(c0, c1, 0,1,2,3,4,5,6,7,8,9,10,11,12,13,14,15);
}

__device__ __forceinline__ v16bf load_tile_b(const unsigned short* __restrict__ base,
                                             int row_base, int k, int lane) {
  const int r  = row_base + (lane & 15);
  const int kc = k + ((lane >> 4) << 3);
  const __bf16* q = (const __bf16*)(base + (size_t)r * HID + kc);
  v8bf c0 = *(const v8bf*)(q);
  v8bf c1 = *(const v8bf*)(q + 16);
  return __builtin_shufflevector(c0, c1, 0,1,2,3,4,5,6,7,8,9,10,11,12,13,14,15);
}

// =====================================================================
// 32x32 register-blocked tile: 4 accumulators, A/B fragments reused 2x.
// Per K-step: 8 v16bf loads -> 12 WMMAs (vs 4 loads -> 3 WMMAs unblocked).
// Split-bf16 fp32 emulation: hi*hi + hi*lo + lo*hi into f32 accumulator.
// acc[0]=(m0,n0) acc[1]=(m0,n1) acc[2]=(m1,n0) acc[3]=(m1,n1)
// =====================================================================
__device__ __forceinline__ void gemm_2x2(const unsigned short* __restrict__ Ahi,
                                         const unsigned short* __restrict__ Alo,
                                         const unsigned short* __restrict__ Bhi,
                                         const unsigned short* __restrict__ Blo,
                                         int arow, int brow, int lane, v8f acc[4]) {
  v8f z = {};
  acc[0] = z; acc[1] = z; acc[2] = z; acc[3] = z;
  #pragma unroll 1
  for (int k = 0; k < HID; k += 32) {
    v16bf a0h = load_tile_a(Ahi, arow,      k, lane);
    v16bf a0l = load_tile_a(Alo, arow,      k, lane);
    v16bf a1h = load_tile_a(Ahi, arow + 16, k, lane);
    v16bf a1l = load_tile_a(Alo, arow + 16, k, lane);
    v16bf b0h = load_tile_b(Bhi, brow,      k, lane);
    v16bf b0l = load_tile_b(Blo, brow,      k, lane);
    v16bf b1h = load_tile_b(Bhi, brow + 16, k, lane);
    v16bf b1l = load_tile_b(Blo, brow + 16, k, lane);
    // interleaved so each accumulator's RAW chain is 4 WMMAs apart
    acc[0] = WMMA_BF16(a0h, b0h, acc[0]);
    acc[1] = WMMA_BF16(a0h, b1h, acc[1]);
    acc[2] = WMMA_BF16(a1h, b0h, acc[2]);
    acc[3] = WMMA_BF16(a1h, b1h, acc[3]);
    acc[0] = WMMA_BF16(a0h, b0l, acc[0]);
    acc[1] = WMMA_BF16(a0h, b1l, acc[1]);
    acc[2] = WMMA_BF16(a1h, b0l, acc[2]);
    acc[3] = WMMA_BF16(a1h, b1l, acc[3]);
    acc[0] = WMMA_BF16(a0l, b0h, acc[0]);
    acc[1] = WMMA_BF16(a0l, b1h, acc[1]);
    acc[2] = WMMA_BF16(a1l, b0h, acc[2]);
    acc[3] = WMMA_BF16(a1l, b1h, acc[3]);
  }
}

// =====================================================================
// 2) GEMM1 + distance epilogue (one wave = 32 batch rows x 32 centers)
//    dist[b, cls] = max over 4 centers of exp(-(|f|^2+|w|^2-2 f.w)/sigma)
// =====================================================================
__device__ __forceinline__ void dist_epilogue(const v8f& acc, int mbase, int nbase,
                                              const float* __restrict__ fsq,
                                              const float* __restrict__ wsq,
                                              float* __restrict__ out, int lane) {
  const int nl = lane & 15;
  const int n  = nbase + nl;                    // center index
  const float wsqn = wsq[n];
  const int mofs = (lane >> 4) << 3;            // C layout: upper half -> M+8
  #pragma unroll
  for (int j = 0; j < 8; ++j) {
    const int b = mbase + j + mofs;             // batch row
    float norm = fsq[b] + wsqn - 2.0f * acc[j];
    float e = __expf(norm * (-1.0f / 10.0f));   // SIGMA = 10
    e = fmaxf(e, __shfl_xor(e, 1));             // max over 4 centers / class
    e = fmaxf(e, __shfl_xor(e, 2));
    if ((lane & 3) == 0) out[(size_t)b * OUTC + (n >> 2)] = e;
  }
}

__global__ __launch_bounds__(256) void gemm1_dist_kernel(const unsigned short* __restrict__ fhi,
                                                         const unsigned short* __restrict__ flo,
                                                         const unsigned short* __restrict__ whi,
                                                         const unsigned short* __restrict__ wlo,
                                                         const float* __restrict__ fsq,
                                                         const float* __restrict__ wsq,
                                                         float* __restrict__ out) {
  const int wid  = threadIdx.x >> 5;
  const int lane = threadIdx.x & 31;
  const int id = blockIdx.x * 8 + wid;          // 0 .. NT2_M*NT2_N-1 (exact)
  const int tm = id / NT2_N;
  const int tn = id - tm * NT2_N;

  v8f acc[4];
  gemm_2x2(fhi, flo, whi, wlo, tm * 32, tn * 32, lane, acc);

  dist_epilogue(acc[0], tm * 32,      tn * 32,      fsq, wsq, out, lane);
  dist_epilogue(acc[1], tm * 32,      tn * 32 + 16, fsq, wsq, out, lane);
  dist_epilogue(acc[2], tm * 32 + 16, tn * 32,      fsq, wsq, out, lane);
  dist_epilogue(acc[3], tm * 32 + 16, tn * 32 + 16, fsq, wsq, out, lane);
}

// =====================================================================
// 3) GEMM2 fused reduce: upper-triangular 32x32 super-tiles of W.W^T,
//    accumulate S1 = sum d, S2 = sum d^2 over strict upper triangle
// =====================================================================
__device__ __forceinline__ void pair_epilogue(const v8f& acc, int mbase, int nbase,
                                              const float* __restrict__ wsq,
                                              double& s1, double& s2, int lane) {
  const int n = nbase + (lane & 15);
  const float wsqn = wsq[n];
  const int mofs = (lane >> 4) << 3;
  #pragma unroll
  for (int j = 0; j < 8; ++j) {
    const int m = mbase + j + mofs;
    float d = wsq[m] + wsqn - 2.0f * acc[j];
    if (m < n) {                                // strict upper triangle
      double dd = (double)d;
      s1 += dd;
      s2 += dd * dd;
    }
  }
}

__global__ __launch_bounds__(256) void gemm2_reduce_kernel(const unsigned short* __restrict__ whi,
                                                           const unsigned short* __restrict__ wlo,
                                                           const float* __restrict__ wsq,
                                                           double* __restrict__ S) {
  const int wid  = threadIdx.x >> 5;
  const int lane = threadIdx.x & 31;
  const int ti = blockIdx.y;
  const int tj = blockIdx.x * 8 + wid;
  if (tj >= NT2_N || tj < ti) return;           // wave-uniform cull (triangle)

  v8f acc[4];
  gemm_2x2(whi, wlo, whi, wlo, ti * 32, tj * 32, lane, acc);

  double s1 = 0.0, s2 = 0.0;
  pair_epilogue(acc[0], ti * 32,      tj * 32,      wsq, s1, s2, lane);
  pair_epilogue(acc[1], ti * 32,      tj * 32 + 16, wsq, s1, s2, lane);
  pair_epilogue(acc[2], ti * 32 + 16, tj * 32,      wsq, s1, s2, lane);
  pair_epilogue(acc[3], ti * 32 + 16, tj * 32 + 16, wsq, s1, s2, lane);

  #pragma unroll
  for (int o = 16; o > 0; o >>= 1) {
    s1 += __shfl_xor(s1, o);
    s2 += __shfl_xor(s2, o);
  }
  if (lane == 0) {
    atomicAdd(&S[0], s1);                       // global_atomic_add_f64
    atomicAdd(&S[1], s2);
  }
}

// =====================================================================
// 4) tiny kernels: zero accumulators / broadcast rw into last column
// =====================================================================
__global__ void zero_acc_kernel(double* S) {
  if (threadIdx.x < 2) S[threadIdx.x] = 0.0;
}

__global__ void write_rw_kernel(const double* __restrict__ S, float* __restrict__ out) {
  const int b = blockIdx.x * blockDim.x + threadIdx.x;
  if (b >= NBATCH) return;
  const double cnt = ((double)MC * (double)MC - (double)MC) * 0.5;   // 7,998,000
  const double s1 = S[0], s2 = S[1];
  const double rw = (s2 - s1 * s1 / cnt) / cnt;  // == denom*(S2 - 2 mu S1 + cnt mu^2)
  out[(size_t)b * OUTC + 1000] = (float)rw;
}

// =====================================================================
// launch
// =====================================================================
extern "C" void kernel_launch(void* const* d_in, const int* in_sizes, int n_in,
                              void* d_out, int out_size, void* d_ws, size_t ws_size,
                              hipStream_t stream) {
  (void)in_sizes; (void)n_in; (void)out_size; (void)ws_size;

  const float* f = (const float*)d_in[0];        // [512, 512]
  const float* W = (const float*)d_in[1];        // [1000, 4, 512] -> [4000, 512]
  float* out = (float*)d_out;                    // [512, 1001]

  // workspace layout (~9.3 MB)
  char* base = (char*)d_ws;
  double*         S    = (double*)(base);                       // 2 doubles
  float*          fsq  = (float*)(base + 256);                  // 512 f32
  float*          wsq  = (float*)(base + 256 + 2048);           // 4000 f32
  unsigned short* f_hi = (unsigned short*)(base + 256 + 2048 + 16128);
  unsigned short* f_lo = f_hi + (size_t)NBATCH * HID;
  unsigned short* w_hi = f_lo + (size_t)NBATCH * HID;
  unsigned short* w_lo = w_hi + (size_t)MC * HID;

  zero_acc_kernel<<<1, 32, 0, stream>>>(S);
  split_kernel<<<NBATCH, 256, 0, stream>>>(f, f_hi, f_lo, fsq);
  split_kernel<<<MC,     256, 0, stream>>>(W, w_hi, w_lo, wsq);

  // 16*125 = 2000 super-tiles (32x32 each), 8 waves / block
  gemm1_dist_kernel<<<(NT2_M * NT2_N) / 8, 256, 0, stream>>>(f_hi, f_lo, w_hi, w_lo,
                                                             fsq, wsq, out);
  // tj = blockIdx.x*8 + wid, ti = blockIdx.y; upper triangle of 125x125 super-tiles
  gemm2_reduce_kernel<<<dim3((NT2_N + 7) / 8, NT2_N), 256, 0, stream>>>(w_hi, w_lo, wsq, S);

  write_rw_kernel<<<2, 256, 0, stream>>>(S, out);
}
